// GNNModel_31138512896531
// MI455X (gfx1250) — matrix operand order repaired and checked by reading, not compiled, not run on hardware
//
#include <hip/hip_runtime.h>
#include <hip/hip_bf16.h>

typedef __attribute__((ext_vector_type(16))) __bf16 v16bf;
typedef __attribute__((ext_vector_type(8)))  float  v8f;

union Frag16 { v16bf v; unsigned u[8]; };

#define HID 128
#define KENC 896   // 768 text + 128 dw
#define KSAGE 256  // 128 self + 128 neigh

// round-to-nearest-even f32 -> bf16 bit pattern
static __device__ __forceinline__ unsigned short f2bf(float f) {
    unsigned u = __builtin_bit_cast(unsigned, f);
    u += 0x7FFFu + ((u >> 16) & 1u);
    return (unsigned short)(u >> 16);
}

// ---------------------------------------------------------------------------
// zero fill
// ---------------------------------------------------------------------------
__global__ __launch_bounds__(256) void zero_kernel(float* __restrict__ p, long n) {
    long i = (long)blockIdx.x * 256 + threadIdx.x;
    if (i < n) p[i] = 0.0f;
}

// ---------------------------------------------------------------------------
// one-time weight prep: transpose + f32->bf16, concatenated K layouts
//   wenc  : [128][896]   = [W_text ; W_dw]^T
//   wsage : [4][128][256] = [W_self(l,d) ; W_neigh(l,d)]^T
// ---------------------------------------------------------------------------
__global__ __launch_bounds__(256) void prep_weights(const float* __restrict__ Wt,
                                                    const float* __restrict__ Wd,
                                                    const float* __restrict__ Wself,
                                                    const float* __restrict__ Wneigh,
                                                    unsigned short* __restrict__ wenc,
                                                    unsigned short* __restrict__ wsage) {
    const long ENC = (long)HID * KENC;           // 114688
    const long SAGE = (long)HID * KSAGE;         // 32768 per (l,dir)
    long i = (long)blockIdx.x * 256 + threadIdx.x;
    if (i < ENC) {
        int n = (int)(i / KENC), k = (int)(i % KENC);
        float v = (k < 768) ? Wt[(size_t)k * HID + n] : Wd[(size_t)(k - 768) * HID + n];
        wenc[i] = f2bf(v);
    } else if (i < ENC + 4 * SAGE) {
        long j = i - ENC;
        int p = (int)(j / SAGE);
        int r = (int)(j % SAGE);
        int n = r / KSAGE, k = r % KSAGE;
        float v = (k < HID) ? Wself[(size_t)p * HID * HID + (size_t)k * HID + n]
                            : Wneigh[(size_t)p * HID * HID + (size_t)(k - HID) * HID + n];
        wsage[j] = f2bf(v);
    }
}

// ---------------------------------------------------------------------------
// degree counting (both directions at once) and inversion
// ---------------------------------------------------------------------------
__global__ __launch_bounds__(256) void deg_kernel(const int* __restrict__ src,
                                                  const int* __restrict__ dst,
                                                  float* __restrict__ degf,
                                                  float* __restrict__ degb, int E) {
    int e = blockIdx.x * 256 + threadIdx.x;
    if (e < E) {
        atomicAdd(&degf[dst[e]], 1.0f);
        atomicAdd(&degb[src[e]], 1.0f);
    }
}

__global__ __launch_bounds__(256) void invdeg_kernel(float* __restrict__ d, long n) {
    long i = (long)blockIdx.x * 256 + threadIdx.x;
    if (i < n) d[i] = 1.0f / fmaxf(d[i], 1.0f);
}

// ---------------------------------------------------------------------------
// LayerNorm: one wave32 per node, 4 channels per lane, shuffle reduction
// ---------------------------------------------------------------------------
__global__ __launch_bounds__(256) void ln_kernel(const float* __restrict__ x,
                                                 float* __restrict__ h,
                                                 const float* __restrict__ g,
                                                 const float* __restrict__ b, int n_nodes) {
    int wave = threadIdx.x >> 5, lane = threadIdx.x & 31;
    int node = blockIdx.x * 8 + wave;
    if (node >= n_nodes) return;
    float4 v = ((const float4*)(x + (size_t)node * HID))[lane];
    float s  = v.x + v.y + v.z + v.w;
    float sq = v.x * v.x + v.y * v.y + v.z * v.z + v.w * v.w;
    #pragma unroll
    for (int off = 16; off > 0; off >>= 1) {
        s  += __shfl_xor(s, off, 32);
        sq += __shfl_xor(sq, off, 32);
    }
    float mu  = s * (1.0f / HID);
    float var = sq * (1.0f / HID) - mu * mu;
    float rs  = rsqrtf(var + 1e-5f);
    float4 gg = ((const float4*)g)[lane];
    float4 bb = ((const float4*)b)[lane];
    float4 o;
    o.x = (v.x - mu) * rs * gg.x + bb.x;
    o.y = (v.y - mu) * rs * gg.y + bb.y;
    o.z = (v.z - mu) * rs * gg.z + bb.z;
    o.w = (v.w - mu) * rs * gg.w + bb.w;
    ((float4*)(h + (size_t)node * HID))[lane] = o;
}

// ---------------------------------------------------------------------------
// edge scatter: neigh[d[e]][c] += h[s[e]][c]; one thread per (edge, channel)
// ---------------------------------------------------------------------------
__global__ __launch_bounds__(256) void scatter_kernel(const float* __restrict__ h,
                                                      const int* __restrict__ sidx,
                                                      const int* __restrict__ didx,
                                                      float* __restrict__ neigh, int E) {
    long i = (long)blockIdx.x * 256 + threadIdx.x;
    if (i >= (long)E * HID) return;
    int e = (int)(i >> 7), c = (int)(i & (HID - 1));
    atomicAdd(&neigh[(long)didx[e] * HID + c], h[(long)sidx[e] * HID + c]);
}

// ---------------------------------------------------------------------------
// bf16 WMMA GEMM skeleton (shared by encoder & sage):
//   block: 256 threads = 8 waves; M-tile 128 rows, each wave owns 16 N cols,
//   8 M-subtiles per wave; A double-buffered through LDS (f32->bf16 convert),
//   B fragments read directly from pre-transposed bf16 global (32B/lane).
// ---------------------------------------------------------------------------

// Encoder GEMM: x = [text|dw] @ wenc^T + (bt+bd)
__global__ __launch_bounds__(256) void encoder_gemm(const float* __restrict__ text,
                                                    const float* __restrict__ dwe,
                                                    const unsigned short* __restrict__ Bt,
                                                    const float* __restrict__ bt,
                                                    const float* __restrict__ bd,
                                                    float* __restrict__ x, int n_nodes) {
    __shared__ unsigned short As[2][128][34];
    const int tid = threadIdx.x;
    const int wave = tid >> 5, lane = tid & 31;
    const int half = lane >> 4, l15 = lane & 15;
    const int row0 = blockIdx.x * 128;
    const int NC = KENC / 32;  // 28

    auto stageA = [&](int kc, int buf) {
        int kk = kc * 32;
        #pragma unroll
        for (int i = tid; i < 128 * 32; i += 256) {
            int r = i >> 5, k = i & 31;
            int gr = row0 + r, gk = kk + k;
            float v = 0.0f;
            if (gr < n_nodes)
                v = (gk < 768) ? text[(size_t)gr * 768 + gk]
                               : dwe[(size_t)gr * HID + (gk - 768)];
            As[buf][r][k] = f2bf(v);
        }
    };

    v8f acc[8];
    #pragma unroll
    for (int m = 0; m < 8; ++m) acc[m] = (v8f)0.0f;

    stageA(0, 0);
    __syncthreads();

    const int nrow = wave * 16 + l15;
    for (int kc = 0; kc < NC; ++kc) {
        if (kc + 1 < NC) stageA(kc + 1, (kc + 1) & 1);  // prefetch next chunk

        // B fragment: 32 contiguous bytes from pre-transposed bf16 weights
        const uint4* bp =
            (const uint4*)(Bt + (size_t)nrow * KENC + kc * 32 + half * 16);
        uint4 b0 = bp[0], b1 = bp[1];
        Frag16 bfr;
        bfr.u[0] = b0.x; bfr.u[1] = b0.y; bfr.u[2] = b0.z; bfr.u[3] = b0.w;
        bfr.u[4] = b1.x; bfr.u[5] = b1.y; bfr.u[6] = b1.z; bfr.u[7] = b1.w;

        const int buf = kc & 1;
        #pragma unroll
        for (int ms = 0; ms < 8; ++ms) {
            Frag16 af;
            int ar = ms * 16 + l15;
            #pragma unroll
            for (int v = 0; v < 4; ++v) {
                af.u[v]     = *(const unsigned*)&As[buf][ar][half * 16 + v * 2];
                af.u[4 + v] = *(const unsigned*)&As[buf][ar][half * 16 + 8 + v * 2];
            }
            acc[ms] = __builtin_amdgcn_wmma_f32_16x16x32_bf16(
                false, af.v, false, bfr.v, (short)0, acc[ms], false, false);
        }
        __syncthreads();
    }

    const int ncol = nrow;
    float bias = bt[ncol] + bd[ncol];
    #pragma unroll
    for (int ms = 0; ms < 8; ++ms)
        #pragma unroll
        for (int r = 0; r < 8; ++r) {
            int gr = row0 + ms * 16 + r + half * 8;
            if (gr < n_nodes) x[(size_t)gr * HID + ncol] = acc[ms][r] + bias;
        }
}

// SAGE GEMM: relu([h | neigh*invdeg] @ wsage^T + b)
// mode 0: y = relu(...)      mode 1: x = x + y + relu(...)  (skip connection)
__global__ __launch_bounds__(256) void sage_gemm(const float* __restrict__ h,
                                                 const float* __restrict__ neigh,
                                                 const float* __restrict__ invdeg,
                                                 const unsigned short* __restrict__ Bt,
                                                 const float* __restrict__ bc,
                                                 float* __restrict__ y,
                                                 float* __restrict__ x,
                                                 int mode, int n_nodes) {
    __shared__ unsigned short As[2][128][34];
    const int tid = threadIdx.x;
    const int wave = tid >> 5, lane = tid & 31;
    const int half = lane >> 4, l15 = lane & 15;
    const int row0 = blockIdx.x * 128;
    const int NC = KSAGE / 32;  // 8

    auto stageA = [&](int kc, int buf) {
        int kk = kc * 32;
        #pragma unroll
        for (int i = tid; i < 128 * 32; i += 256) {
            int r = i >> 5, k = i & 31;
            int gr = row0 + r, gk = kk + k;
            float v = 0.0f;
            if (gr < n_nodes) {
                if (gk < HID) v = h[(size_t)gr * HID + gk];
                else          v = neigh[(size_t)gr * HID + (gk - HID)] * invdeg[gr];
            }
            As[buf][r][k] = f2bf(v);
        }
    };

    v8f acc[8];
    #pragma unroll
    for (int m = 0; m < 8; ++m) acc[m] = (v8f)0.0f;

    stageA(0, 0);
    __syncthreads();

    const int nrow = wave * 16 + l15;
    for (int kc = 0; kc < NC; ++kc) {
        if (kc + 1 < NC) stageA(kc + 1, (kc + 1) & 1);

        const uint4* bp =
            (const uint4*)(Bt + (size_t)nrow * KSAGE + kc * 32 + half * 16);
        uint4 b0 = bp[0], b1 = bp[1];
        Frag16 bfr;
        bfr.u[0] = b0.x; bfr.u[1] = b0.y; bfr.u[2] = b0.z; bfr.u[3] = b0.w;
        bfr.u[4] = b1.x; bfr.u[5] = b1.y; bfr.u[6] = b1.z; bfr.u[7] = b1.w;

        const int buf = kc & 1;
        #pragma unroll
        for (int ms = 0; ms < 8; ++ms) {
            Frag16 af;
            int ar = ms * 16 + l15;
            #pragma unroll
            for (int v = 0; v < 4; ++v) {
                af.u[v]     = *(const unsigned*)&As[buf][ar][half * 16 + v * 2];
                af.u[4 + v] = *(const unsigned*)&As[buf][ar][half * 16 + 8 + v * 2];
            }
            acc[ms] = __builtin_amdgcn_wmma_f32_16x16x32_bf16(
                false, af.v, false, bfr.v, (short)0, acc[ms], false, false);
        }
        __syncthreads();
    }

    const int ncol = nrow;
    float bias = bc[ncol];
    #pragma unroll
    for (int ms = 0; ms < 8; ++ms)
        #pragma unroll
        for (int r = 0; r < 8; ++r) {
            int gr = row0 + ms * 16 + r + half * 8;
            if (gr < n_nodes) {
                size_t idx = (size_t)gr * HID + ncol;
                float val = fmaxf(acc[ms][r] + bias, 0.0f);
                if (mode == 0) y[idx] = val;
                else           x[idx] = x[idx] + y[idx] + val;
            }
        }
}

// ---------------------------------------------------------------------------
// gather queries + L2 normalize: one wave32 per query row
// ---------------------------------------------------------------------------
__global__ __launch_bounds__(256) void norm_kernel(const float* __restrict__ x,
                                                   const int* __restrict__ qidx,
                                                   float* __restrict__ out, int nq) {
    int wave = threadIdx.x >> 5, lane = threadIdx.x & 31;
    int q = blockIdx.x * 8 + wave;
    if (q >= nq) return;
    int node = qidx[q];
    float4 v = ((const float4*)(x + (size_t)node * HID))[lane];
    float ss = v.x * v.x + v.y * v.y + v.z * v.z + v.w * v.w;
    #pragma unroll
    for (int off = 16; off > 0; off >>= 1) ss += __shfl_xor(ss, off, 32);
    float rs = rsqrtf(ss);
    float4 o = {v.x * rs, v.y * rs, v.z * rs, v.w * rs};
    ((float4*)(out + (size_t)q * HID))[lane] = o;
}

// ---------------------------------------------------------------------------
extern "C" void kernel_launch(void* const* d_in, const int* in_sizes, int n_in,
                              void* d_out, int out_size, void* d_ws, size_t ws_size,
                              hipStream_t stream) {
    const int*   src  = (const int*)d_in[0];
    const int*   dst  = (const int*)d_in[1];
    const float* text = (const float*)d_in[2];
    const float* dwe  = (const float*)d_in[3];
    const int*   qix  = (const int*)d_in[4];
    const float* Wt   = (const float*)d_in[5];
    const float* bt   = (const float*)d_in[6];
    const float* Wd   = (const float*)d_in[7];
    const float* bd   = (const float*)d_in[8];
    const float* lng  = (const float*)d_in[9];
    const float* lnb  = (const float*)d_in[10];
    const float* Wself  = (const float*)d_in[11];
    const float* Wneigh = (const float*)d_in[12];
    const float* bconv  = (const float*)d_in[13];

    const int E  = in_sizes[0];
    const int N  = in_sizes[2] / 768;
    const int NQ = in_sizes[4];

    float* ws    = (float*)d_ws;
    float* x     = ws;
    float* h     = x + (size_t)N * HID;
    float* neigh = h + (size_t)N * HID;
    float* y     = neigh + (size_t)N * HID;
    float* degf  = y + (size_t)N * HID;   // becomes invdeg (fwd: per-dst)
    float* degb  = degf + N;              // becomes invdeg (bwd: per-src)
    size_t fbytes = ((size_t)4 * N * HID + 2 * (size_t)N) * sizeof(float);
    fbytes = (fbytes + 15) & ~(size_t)15;           // 16B align for uint4 B loads
    unsigned short* wenc  = (unsigned short*)((char*)d_ws + fbytes);
    unsigned short* wsage = wenc + (size_t)HID * KENC;

    const int mblocks  = (N + 127) / 128;
    const int nodeW    = (N + 7) / 8;
    const long featN   = (long)N * HID;
    const int featBlk  = (int)((featN + 255) / 256);
    const long edgeN   = (long)E * HID;
    const int edgeBlk  = (int)((edgeN + 255) / 256);

    // one-time weight transpose/convert + degrees -> inverse
    const long wtot = (long)HID * KENC + 4L * HID * KSAGE;
    prep_weights<<<(int)((wtot + 255) / 256), 256, 0, stream>>>(Wt, Wd, Wself, Wneigh,
                                                                wenc, wsage);
    zero_kernel<<<(2 * N + 255) / 256, 256, 0, stream>>>(degf, 2L * N);
    deg_kernel<<<(E + 255) / 256, 256, 0, stream>>>(src, dst, degf, degb, E);
    invdeg_kernel<<<(2 * N + 255) / 256, 256, 0, stream>>>(degf, 2L * N);

    // encoder: x = text@Wt + dw@Wd + bias
    encoder_gemm<<<mblocks, 256, 0, stream>>>(text, dwe, wenc, bt, bd, x, N);

    for (int l = 0; l < 2; ++l) {
        for (int dir = 0; dir < 2; ++dir) {
            int p = l * 2 + dir;
            const int* s = dir ? dst : src;
            const int* d = dir ? src : dst;
            const float* invd = dir ? degb : degf;
            ln_kernel<<<nodeW, 256, 0, stream>>>(x, h, lng + (size_t)p * HID,
                                                 lnb + (size_t)p * HID, N);
            zero_kernel<<<featBlk, 256, 0, stream>>>(neigh, featN);
            scatter_kernel<<<edgeBlk, 256, 0, stream>>>(h, s, d, neigh, E);
            sage_gemm<<<mblocks, 256, 0, stream>>>(
                h, neigh, invd, wsage + (size_t)p * HID * KSAGE,
                bconv + (size_t)p * HID, y, x, dir, N);
        }
    }

    norm_kernel<<<(NQ + 7) / 8, 256, 0, stream>>>(x, qix, (float*)d_out, NQ);
}
